// TSI_Encoder_1494648619478
// MI455X (gfx1250) — matrix operand order repaired and checked by reading, not compiled, not run on hardware
//
#include <hip/hip_runtime.h>
#include <hip/hip_bf16.h>

typedef _Float16 h16;
typedef __attribute__((ext_vector_type(16))) _Float16 v16h;
typedef __attribute__((ext_vector_type(8)))  _Float16 v8h;
typedef __attribute__((ext_vector_type(8)))  float    v8f;
typedef __attribute__((ext_vector_type(4)))  unsigned int v4u;
typedef __attribute__((ext_vector_type(8)))  int v8i;
typedef __attribute__((ext_vector_type(4)))  int v4i;

#define BDIM 512              // 16 waves: 8 M-blocks x 2 N-groups
static constexpr int Bsz  = 2048;
static constexpr int Lsz  = 168;
static constexpr int NBc  = 14;
static constexpr int BSc  = 12;
static constexpr int K0   = 588;
static constexpr int K0P  = 608;   // padded to multiple of 32
static constexpr int N0   = 512;
static constexpr int N1   = 384;
static constexpr int N2   = 256;
static constexpr int MT   = 128;   // batch rows per workgroup
static constexpr int NPOS = 144;
static constexpr int NTOT = Bsz * Lsz; // 344064

// bank-conflict-free LDS row strides (halves)
static constexpr int KA0  = 616;   // A0 stride  (1232 B)
static constexpr int KH0  = 520;   // H0 stride  (1040 B)
static constexpr int KH1  = 392;   // H1 stride  ( 784 B)
static constexpr int BROW = 40;    // B stage row stride (80 B = 64 B data + 16 B pad)
static constexpr int ACTH = MT * KA0;  // 78848 halves: max(A0, H0, H1); sequentially live

#if defined(__has_builtin)
#  if __has_builtin(__builtin_amdgcn_tensor_load_to_lds) && __has_builtin(__builtin_amdgcn_s_wait_tensorcnt)
#    define USE_TDM 1
#  else
#    define USE_TDM 0
#  endif
#else
#  define USE_TDM 0
#endif

// ---------------- stats / prep kernels ----------------

__global__ void reduce_partial(const float* __restrict__ x, float* __restrict__ part, int n) {
    __shared__ float s1[256], s2[256];
    float a = 0.f, b = 0.f;
    for (int i = blockIdx.x * 256 + threadIdx.x; i < n; i += 256 * 256) {
        float v = x[i];
        a += v; b += v * v;
    }
    s1[threadIdx.x] = a; s2[threadIdx.x] = b;
    __syncthreads();
    for (int off = 128; off > 0; off >>= 1) {
        if (threadIdx.x < off) {
            s1[threadIdx.x] += s1[threadIdx.x + off];
            s2[threadIdx.x] += s2[threadIdx.x + off];
        }
        __syncthreads();
    }
    if (threadIdx.x == 0) {
        part[blockIdx.x]       = s1[0];
        part[256 + blockIdx.x] = s2[0];
    }
}

__global__ void finalize_stats(const float* __restrict__ part, float* __restrict__ stats,
                               const float* __restrict__ wmask, float* __restrict__ maskout, int n) {
    __shared__ float s1[256], s2[256];
    s1[threadIdx.x] = part[threadIdx.x];
    s2[threadIdx.x] = part[256 + threadIdx.x];
    __syncthreads();
    for (int off = 128; off > 0; off >>= 1) {
        if (threadIdx.x < off) {
            s1[threadIdx.x] += s1[threadIdx.x + off];
            s2[threadIdx.x] += s2[threadIdx.x + off];
        }
        __syncthreads();
    }
    if (threadIdx.x == 0) {
        float S = s1[0], S2 = s2[0];
        float mean = S / (float)n;
        float var  = (S2 - S * S / (float)n) / (float)(n - 1); // ddof=1
        stats[0] = mean;
        stats[1] = rsqrtf(var);
    }
    if (threadIdx.x < 16) {
        int g = threadIdx.x;
        float mx = -1e30f;
        for (int j = 0; j < NPOS; ++j) mx = fmaxf(mx, wmask[g * NPOS + j]);
        float ssum = 0.f;
        for (int j = 0; j < NPOS; ++j) ssum += expf(wmask[g * NPOS + j] - mx);
        float inv = 1.f / ssum;
        for (int j = 0; j < NPOS; ++j)
            maskout[g * NPOS + j] = expf(wmask[g * NPOS + j] - mx) * inv;
    }
}

__global__ void normalize_x(const float* __restrict__ x, const float* __restrict__ stats,
                            float* __restrict__ xn, int n) {
    int i = blockIdx.x * 256 + threadIdx.x;
    if (i < n) xn[i] = (x[i] - stats[0]) * stats[1];
}

__global__ void convert_pad(const float* __restrict__ src, h16* __restrict__ dst,
                            int N, int K, int Kpad) {
    int idx = blockIdx.x * 256 + threadIdx.x;
    if (idx >= N * Kpad) return;
    int n = idx / Kpad, k = idx % Kpad;
    dst[idx] = (k < K) ? (h16)src[n * K + k] : (h16)0.f;
}

__global__ void zero_out(float* __restrict__ p, int n) {
    int i = blockIdx.x * 256 + threadIdx.x;
    if (i < n) p[i] = 0.f;
}

__global__ void relu_out(float* __restrict__ p, int n) {
    int i = blockIdx.x * 256 + threadIdx.x;
    if (i < n) p[i] = fmaxf(p[i], 0.f);
}

// ---------------- TDM helpers ----------------

__device__ __forceinline__ unsigned lds_byte_addr(const void* p) {
    return (unsigned)(uintptr_t)(__attribute__((address_space(3))) const void*)p;
}

#if USE_TDM
// 2D tile DMA: tile_dim0 = 32 elems (contiguous K, 64B), tile_dim1 = nrows,
// elem 2B, global row stride = kk elems. LDS dest: 64B row + 16B pad -> 80B stride
// (pad_interval=3: every 16 DWORDs; pad_amount=3: 4 DWORDs).
__device__ __forceinline__ void tdm_load_slice(const h16* gsrc, unsigned lds_addr,
                                               int nrows, int kk) {
    unsigned long long ga = (unsigned long long)(uintptr_t)gsrc;
    v4u g0;
    g0[0] = 1u;                                   // count=1 (valid), load, user mode
    g0[1] = lds_addr;                             // D#.lds_addr
    g0[2] = (unsigned)(ga & 0xffffffffu);         // global_addr[31:0]
    g0[3] = (unsigned)((ga >> 32) & 0x01ffffffu)  // global_addr[56:32]
          | (2u << 30);                           // type=2 ("image")
    v8i g1;
    g1[0] = (int)((1u << 16)                      // data_size=1 -> 2 bytes
          | (1u << 20)                            // pad_enable
          | (3u << 22)                            // pad_interval: every 16 DWORDs (64B)
          | (3u << 25));                          // pad_amount: 4 DWORDs (16B)
    g1[1] = (int)(((unsigned)kk & 0xffffu) << 16);                // tensor_dim0[15:0]
    g1[2] = (int)((((unsigned)kk >> 16) & 0xffffu)
          | (((unsigned)nrows & 0xffffu) << 16));                 // dim0[31:16] | dim1[15:0]
    g1[3] = (int)((((unsigned)nrows >> 16) & 0xffffu)
          | (32u << 16));                                         // dim1[31:16] | tile_dim0=32
    g1[4] = (int)((unsigned)nrows & 0xffffu);                     // tile_dim1 = nrows, tile_dim2=0
    g1[5] = kk;                                                   // tensor_dim0_stride[31:0]
    g1[6] = 0;
    g1[7] = 0;
    v4i z4 = {0, 0, 0, 0};
    v8i z8 = {0, 0, 0, 0, 0, 0, 0, 0};
    __builtin_amdgcn_tensor_load_to_lds(g0, g1, z4, z4, z8, 0);
}
#endif

// ---------------- fused MLP (WMMA) ----------------
// KK: weight K (global row stride), KAS: A-tile LDS row stride,
// NOUT: output channels, HSTR: H store stride, NT: n-tiles per wave.
template<int KK, int KAS, int NOUT, int HSTR, int NT>
__device__ __forceinline__ void gemm_layer(
    const h16* __restrict__ As, const h16* __restrict__ Wg,
    const float* __restrict__ bias, h16* __restrict__ Bst,  // 2 buffers of N0*BROW halves
    h16* __restrict__ Hd,                 // next-layer activations (null for final)
    float* __restrict__ outp,             // final output (atomic accum)
    const float* __restrict__ maskp,      // [g*144 + hw]
    int hw, int rowbase, int wv, int lane, int mb, int ng)
{
    const int t      = threadIdx.x;
    const int laneN  = lane & 15;
    const int laneHi = lane >> 4;
    constexpr int KS   = KK / 32;
    constexpr int BUFH = N0 * BROW;  // halves per B buffer

    v8f acc[NT] = {};

#if USE_TDM
    if (wv == 0) {
        tdm_load_slice(Wg, lds_byte_addr(Bst), NOUT, KK);
        __builtin_amdgcn_s_wait_tensorcnt(0);
    }
    __syncthreads();
    for (int ks = 0; ks < KS; ++ks) {
        const h16* Bbuf = Bst + (ks & 1) * BUFH;
        if (wv == 0 && ks + 1 < KS)
            tdm_load_slice(Wg + (ks + 1) * 32,
                           lds_byte_addr(Bst + ((ks + 1) & 1) * BUFH), NOUT, KK);

        const h16* ap = As + (mb * 16 + laneN) * KAS + ks * 32 + laneHi * 8;
        v8h alo = *(const v8h*)ap;
        v8h ahi = *(const v8h*)(ap + 16);
        v16h afr;
#pragma unroll
        for (int i = 0; i < 8; ++i) { afr[i] = alo[i]; afr[8 + i] = ahi[i]; }

        const h16* bpBase = Bbuf + (ng * NT * 16 + laneN) * BROW + laneHi * 8;
        auto loadB = [&](int ti) -> v16h {
            const h16* bp = bpBase + ti * 16 * BROW;
            v8h lo = *(const v8h*)bp;
            v8h hi = *(const v8h*)(bp + 16);
            v16h r;
#pragma unroll
            for (int i = 0; i < 8; ++i) { r[i] = lo[i]; r[8 + i] = hi[i]; }
            return r;
        };
        v16h bcur = loadB(0);
        v16h bnx  = (NT > 1) ? loadB(1) : bcur;
#pragma unroll
        for (int ti = 0; ti < NT; ++ti) {
            v16h b2 = bcur;
            if (ti + 2 < NT) b2 = loadB(ti + 2);
            acc[ti] = __builtin_amdgcn_wmma_f32_16x16x32_f16(
                false, afr, false, bcur, (short)0, acc[ti], false, false);
            bcur = bnx;
            bnx  = b2;
        }
        if (wv == 0 && ks + 1 < KS) __builtin_amdgcn_s_wait_tensorcnt(0);
        __syncthreads();
    }
#else
    for (int ks = 0; ks < KS; ++ks) {
        __syncthreads();
        for (int r = t; r < NOUT; r += BDIM) {
            const uint4* src = (const uint4*)(Wg + r * KK + ks * 32);
            uint4* dst = (uint4*)(Bst + r * BROW);
            dst[0] = src[0]; dst[1] = src[1]; dst[2] = src[2]; dst[3] = src[3];
        }
        __syncthreads();
        const h16* ap = As + (mb * 16 + laneN) * KAS + ks * 32 + laneHi * 8;
        v8h alo = *(const v8h*)ap;
        v8h ahi = *(const v8h*)(ap + 16);
        v16h afr;
#pragma unroll
        for (int i = 0; i < 8; ++i) { afr[i] = alo[i]; afr[8 + i] = ahi[i]; }
        const h16* bpBase = Bst + (ng * NT * 16 + laneN) * BROW + laneHi * 8;
        auto loadB = [&](int ti) -> v16h {
            const h16* bp = bpBase + ti * 16 * BROW;
            v8h lo = *(const v8h*)bp;
            v8h hi = *(const v8h*)(bp + 16);
            v16h r;
#pragma unroll
            for (int i = 0; i < 8; ++i) { r[i] = lo[i]; r[8 + i] = hi[i]; }
            return r;
        };
        v16h bcur = loadB(0);
        v16h bnx  = (NT > 1) ? loadB(1) : bcur;
#pragma unroll
        for (int ti = 0; ti < NT; ++ti) {
            v16h b2 = bcur;
            if (ti + 2 < NT) b2 = loadB(ti + 2);
            acc[ti] = __builtin_amdgcn_wmma_f32_16x16x32_f16(
                false, afr, false, bcur, (short)0, acc[ti], false, false);
            bcur = bnx;
            bnx  = b2;
        }
    }
    __syncthreads();
#endif

    if (Hd != nullptr) {
#pragma unroll
        for (int ti = 0; ti < NT; ++ti) {
            int nt = ng * NT + ti;
            int N = nt * 16 + laneN;
            float bv = bias[N];
#pragma unroll
            for (int r = 0; r < 8; ++r) {
                int M = mb * 16 + (laneHi ? (r + 8) : r);
                float v = fmaxf(acc[ti][r] + bv, 0.f);
                Hd[M * HSTR + N] = (h16)v;
            }
        }
    } else {
#pragma unroll
        for (int ti = 0; ti < NT; ++ti) {
            int nt = ng * NT + ti;
            int N = nt * 16 + laneN;
            float bv = bias[N];
            float mv = maskp[(N >> 4) * NPOS + hw];
#pragma unroll
            for (int r = 0; r < 8; ++r) {
                int M = mb * 16 + (laneHi ? (r + 8) : r);
                float v = fmaxf(acc[ti][r] + bv, 0.f);
                atomicAdd(&outp[(rowbase + M) * N2 + N], v * mv);
            }
        }
    }
    __syncthreads();
}

__global__ __launch_bounds__(BDIM) void mlp_fused(
    const float* __restrict__ xn,
    const h16* __restrict__ w0h, const float* __restrict__ b0v,
    const h16* __restrict__ w1h, const float* __restrict__ b1v,
    const h16* __restrict__ w2h, const float* __restrict__ b2v,
    const float* __restrict__ maskf, float* __restrict__ out)
{
    // ACT holds A0, then H0, then H1 (each dead before the next is written;
    // transitions are separated by full-workgroup barriers at K-loop ends)
    __shared__ __align__(64) h16 ACT[ACTH];           // 157696 B
    __shared__ __align__(64) h16 Bst[2 * N0 * BROW];  //  81920 B (double buffered, padded rows)
    __shared__ float XP[MT * NBc];                    //   7168 B
    __shared__ float XQ[MT * NBc];                    //   7168 B

    const int t    = threadIdx.x;
    const int wv   = t >> 5;              // 0..15
    const int lane = t & 31;
    const int mb   = wv & 7;              // 8 M blocks of 16 rows
    const int ng   = wv >> 3;             // 2 N groups
    const int hw   = blockIdx.x;          // 0..143
    const int hh   = hw / BSc, ww = hw % BSc;
    const int rowbase = blockIdx.y * MT;  // batch base

    for (int i = t; i < MT * NBc; i += BDIM) {
        int m = i / NBc, blk = i % NBc;
        int b = rowbase + m;
        XP[i] = xn[b * Lsz + blk * BSc + hh];  // p = i*12 + h
        XQ[i] = xn[b * Lsz + blk * BSc + ww];  // q = j*12 + w
    }
    __syncthreads();

    for (int idx = t; idx < MT * NBc * NBc; idx += BDIM) {
        int m  = idx / (NBc * NBc);
        int ij = idx % (NBc * NBc);
        int i  = ij / NBc, j = ij % NBc;
        float xp = XP[m * NBc + i];
        float xq = XQ[m * NBc + j];
        float s   = xq + xp + 1e-5f;
        float di  = xq - xp;
        float inv = 1.0f / s;
        h16* dst = &ACT[m * KA0 + ij * 3];
        dst[0] = (h16)di;
        dst[1] = (h16)(di * inv);
        dst[2] = (h16)(xq * inv);
    }
    for (int idx = t; idx < MT * (K0P - K0); idx += BDIM) {
        int m = idx / (K0P - K0), k = idx % (K0P - K0);
        ACT[m * KA0 + K0 + k] = (h16)0.f;
    }
    __syncthreads();

    gemm_layer<K0P, KA0, N0, KH0, 16>(ACT, w0h, b0v, Bst, ACT, nullptr, nullptr, hw, rowbase, wv, lane, mb, ng);
    gemm_layer<N0,  KH0, N1, KH1, 12>(ACT, w1h, b1v, Bst, ACT, nullptr, nullptr, hw, rowbase, wv, lane, mb, ng);
    gemm_layer<N1,  KH1, N2, 0,    8>(ACT, w2h, b2v, Bst, nullptr, out, maskf, hw, rowbase, wv, lane, mb, ng);
}

// ---------------- launcher ----------------

extern "C" void kernel_launch(void* const* d_in, const int* in_sizes, int n_in,
                              void* d_out, int out_size, void* d_ws, size_t ws_size,
                              hipStream_t stream) {
    const float* x  = (const float*)d_in[0];
    const float* w0 = (const float*)d_in[1];
    const float* b0 = (const float*)d_in[2];
    const float* w1 = (const float*)d_in[3];
    const float* b1 = (const float*)d_in[4];
    const float* w2 = (const float*)d_in[5];
    const float* b2 = (const float*)d_in[6];
    const float* wm = (const float*)d_in[7];
    float* out = (float*)d_out;

    float* wsf   = (float*)d_ws;
    float* part  = wsf;              // 512 floats
    float* stats = wsf + 512;        // 2 floats
    float* maskb = wsf + 520;        // 2304 floats
    float* xn    = wsf + 2880;       // 344064 floats
    h16*   w0h   = (h16*)(wsf + 2880 + NTOT);   // 512*608 halves
    h16*   w1h   = w0h + 512 * K0P;             // 384*512
    h16*   w2h   = w1h + 384 * 512;             // 256*384

    reduce_partial<<<256, 256, 0, stream>>>(x, part, NTOT);
    finalize_stats<<<1, 256, 0, stream>>>(part, stats, wm, maskb, NTOT);
    normalize_x<<<(NTOT + 255) / 256, 256, 0, stream>>>(x, stats, xn, NTOT);
    convert_pad<<<(512 * K0P + 255) / 256, 256, 0, stream>>>(w0, w0h, 512, K0, K0P);
    convert_pad<<<(384 * 512 + 255) / 256, 256, 0, stream>>>(w1, w1h, 384, 512, 512);
    convert_pad<<<(256 * 384 + 255) / 256, 256, 0, stream>>>(w2, w2h, 256, 384, 384);
    zero_out<<<(Bsz * N2 + 255) / 256, 256, 0, stream>>>(out, Bsz * N2);

    dim3 grid(NPOS, Bsz / MT);  // 144 x 16
    mlp_fused<<<grid, BDIM, 0, stream>>>(xn, w0h, b0, w1h, b1, w2h, b2, maskb, out);

    relu_out<<<(Bsz * N2 + 255) / 256, 256, 0, stream>>>(out, Bsz * N2);
}
